// SynthesisDecoder_58703613002457
// MI455X (gfx1250) — matrix operand order repaired and checked by reading, not compiled, not run on hardware
//
#include <hip/hip_runtime.h>
#include <hip/hip_bf16.h>
#include <math.h>

// ---------------------------------------------------------------------------
// SynthesisDecoder for MI455X (gfx1250, wave32, WMMA).
//
// All convolutions are WMMA GEMMs (v_wmma_f32_16x16x32_f16, f32 accum).
// Activations live in packed f16 pixel-major [b][y][x][c] (c padded to 32)
// so B fragments are contiguous b128 loads; weights are pre-packed into
// exact A-fragment layout so A is one aligned v16h load.  Each wave computes
// a 16(M) x 32(N) output tile: two WMMAs per K-step share one A fragment.
// Conv padding is branch-free (clamped address + 0/1 f16 mask multiply) so
// EXEC stays all-ones throughout the K loop.  global_prefetch_b8 hides the
// next K-step's latency.
// ---------------------------------------------------------------------------

typedef __attribute__((ext_vector_type(16))) _Float16 v16h;
typedef __attribute__((ext_vector_type(8)))  _Float16 v8h;
typedef __attribute__((ext_vector_type(8)))  float    v8f;

#define ACT_NONE    0
#define ACT_LRELU   1
#define ACT_GELU    2
#define ACT_SIGMOID 3

// ---------------------------------------------------------------------------
// Weight pre-pack: OIHW f32 -> A-fragment f16 layout
//   apk[((s*nMt + mt)*32 + lane)*16 + j]
// K order = (kh, kw, ci) with ci padded to Cpg (mult of 32); zero-fill pads.
// A lane layout: row = mt*16 + (lane&15);
//   K = s*32 + (lane>>4)*16 + j for j = 0..15 (contiguous, same order as the B loads)
// ---------------------------------------------------------------------------
__global__ void __launch_bounds__(256)
pack_a_kernel(const float* __restrict__ w, _Float16* __restrict__ apk,
              int Cout, int cg, int Cpg, int KK, int nMt, int S)
{
    int t = blockIdx.x * blockDim.x + threadIdx.x;
    int total = S * nMt * 32;
    if (t >= total) return;
    int lane = t & 31;
    int mt   = (t >> 5) % nMt;
    int s    = t / (32 * nMt);
    int hi   = lane >> 4;
    int m    = mt * 16 + (lane & 15);
    v16h frag;
#pragma unroll
    for (int j = 0; j < 16; ++j) {
        int kk  = s * 32 + hi * 16 + j;
        int tap = kk / Cpg;
        int ci  = kk - tap * Cpg;
        float v = 0.0f;
        if (m < Cout && ci < cg) v = w[((long long)m * cg + ci) * KK + tap];
        frag[j] = (_Float16)v;
    }
    ((v16h*)apk)[t] = frag;
}

// ---------------------------------------------------------------------------
// WMMA conv:  out = act(conv(xpk, apk) + bias) [+ respk]
//   xpk   : packed f16 [B*H*W][CinPad]
//   outpk : optional packed f16 [B*H*W][CoutPad]   (Cout must be mult of 16)
//   out32 : optional f32 NCHW (B, Cout, H, W)
//   respk : optional packed f16 [B*H*W][CoutPad]
// One wave = 16(M) x 32(N); block = 8 waves = 32(M) x 128(N).
// ---------------------------------------------------------------------------
__global__ void __launch_bounds__(256)
wmma_conv_pk(const _Float16* __restrict__ xpk, const _Float16* __restrict__ apk,
             const float* __restrict__ bias, const _Float16* __restrict__ respk,
             _Float16* __restrict__ outpk, float* __restrict__ out32,
             int Bn, int CinPad, int Hh, int Ww,
             int Cout, int CoutPad, int Ksz, int groups, int cg, int act)
{
    const int lane = threadIdx.x & 31;
    const int wave = threadIdx.x >> 5;
    const int wm = wave >> 2;                 // 0..1
    const int wn = wave & 3;                  // 0..3
    const int nMt = (Cout + 15) >> 4;
    const int mt  = blockIdx.y * 2 + wm;
    const long long HW = (long long)Hh * Ww;
    const long long Np = (long long)Bn * HW;
    const long long n0 = (long long)blockIdx.x * 128 + wn * 32;
    if (mt >= nMt || n0 >= Np) return;        // wave-uniform exit

    const int m0   = mt * 16;
    const int Cpg  = (cg + 31) & ~31;         // padded channels per group
    const int KK   = Ksz * Ksz;
    const int S    = (KK * Cpg) >> 5;         // number of 32-wide K steps
    const int pad  = Ksz >> 1;
    const int og   = Cout / groups;
    const int cin0 = (m0 / og) * cg;          // group channel base
    const int hi   = lane >> 4;
    const int cb0  = cin0 + hi * 16;          // lane's B channel base offset

    // lane's two output pixels (N columns lane&15 and lane&15 + 16)
    long long nIdx[2];
    bool nvalid[2];
    int bb[2], yo[2], xo[2];
#pragma unroll
    for (int u = 0; u < 2; ++u) {
        nIdx[u] = n0 + u * 16 + (lane & 15);
        nvalid[u] = (nIdx[u] < Np);
        long long t = nvalid[u] ? nIdx[u] : 0;
        bb[u] = (int)(t / HW); t -= (long long)bb[u] * HW;
        yo[u] = (int)(t / Ww);
        xo[u] = (int)(t - (long long)yo[u] * Ww);
    }

    const v16h* __restrict__ ap = (const v16h*)apk + (long long)mt * 32 + lane;
    const long long astep = (long long)nMt * 32;

    v8f acc0 = {}, acc1 = {};
    int kh = 0, kw = 0, cik = 0;
    for (int s = 0; s < S; ++s) {
        // ---- A fragment: one aligned 32B load, shared by both N tiles ----
        const v16h afrag = ap[(long long)s * astep];
        if (s + 1 < S)
            __builtin_prefetch((const void*)(ap + (long long)(s + 1) * astep), 0, 3);
        // ---- B fragments: contiguous 32B, clamped address + 0/1 mask ----
        v16h bfrag[2];
#pragma unroll
        for (int u = 0; u < 2; ++u) {
            const int iy = yo[u] + kh - pad;
            const int ix = xo[u] + kw - pad;
            const bool oob = (iy < 0) | (iy >= Hh) | (ix < 0) | (ix >= Ww) | !nvalid[u];
            const int iyc = min(max(iy, 0), Hh - 1);
            const int ixc = min(max(ix, 0), Ww - 1);
            const long long pix = ((long long)bb[u] * Hh + iyc) * Ww + ixc;
            const _Float16* bptr = xpk + pix * CinPad + cb0 + cik;
            v16h bf = *(const v16h*)bptr;
            __builtin_prefetch((const void*)(bptr + CinPad), 0, 3);
            const _Float16 mval = oob ? (_Float16)0.0f : (_Float16)1.0f;
            bfrag[u] = bf * mval;
        }
        acc0 = __builtin_amdgcn_wmma_f32_16x16x32_f16(
                   false, afrag, false, bfrag[0], (short)0, acc0, false, false);
        acc1 = __builtin_amdgcn_wmma_f32_16x16x32_f16(
                   false, afrag, false, bfrag[1], (short)0, acc1, false, false);
        cik += 32;
        if (cik >= Cpg) { cik = 0; ++kw; if (kw >= Ksz) { kw = 0; ++kh; } }
    }

    // ---- epilogue: bias + activation (+ packed residual), dual store ----
    v8f accs[2] = { acc0, acc1 };
#pragma unroll
    for (int u = 0; u < 2; ++u) {
        float vals[8];
#pragma unroll
        for (int v = 0; v < 8; ++v) {
            const int m = m0 + hi * 8 + v;
            float val = accs[u][v] + bias[min(m, Cout - 1)];   // branch-free load
            if (act == ACT_LRELU)        val = (val >= 0.0f) ? val : 0.1f * val;
            else if (act == ACT_GELU)    val = 0.5f * val * (1.0f + erff(val * 0.70710678118654752f));
            else if (act == ACT_SIGMOID) val = 1.0f / (1.0f + expf(-val));
            if (respk) val += (float)respk[nIdx[u] * CoutPad + m];
            vals[v] = val;
        }
        if (outpk && nvalid[u]) {       // Cout mult of 16 guaranteed by caller
            v8h ov;
#pragma unroll
            for (int v = 0; v < 8; ++v) ov[v] = (_Float16)vals[v];
            *(v8h*)(outpk + nIdx[u] * CoutPad + m0 + hi * 8) = ov;
        }
        if (out32 && nvalid[u]) {
#pragma unroll
            for (int v = 0; v < 8; ++v) {
                const int m = m0 + hi * 8 + v;
                if (m < Cout)
                    out32[((long long)bb[u] * Cout + m) * HW
                          + (long long)yo[u] * Ww + xo[u]] = vals[v];
            }
        }
    }
}

// ---------------------------------------------------------------------------
// Bilinear resize f32->f32 (half-pixel, clamp), used for flow downsampling.
// ---------------------------------------------------------------------------
__global__ void __launch_bounds__(256)
resize_f32_kernel(const float* __restrict__ src, float* __restrict__ dst,
                  int Bn, int CH, int Hs, int Ws, int Hd, int Wd, float scale)
{
    long long total = (long long)Bn * CH * Hd * Wd;
    long long tid = (long long)blockIdx.x * blockDim.x + threadIdx.x;
    if (tid >= total) return;
    int xd = (int)(tid % Wd);
    int yd = (int)((tid / Wd) % Hd);
    int c  = (int)((tid / ((long long)Wd * Hd)) % CH);
    int b  = (int)(tid / ((long long)CH * Wd * Hd));
    float sy = ((float)yd + 0.5f) * ((float)Hs / (float)Hd) - 0.5f;
    float sx = ((float)xd + 0.5f) * ((float)Ws / (float)Wd) - 0.5f;
    sy = fminf(fmaxf(sy, 0.0f), (float)(Hs - 1));
    sx = fminf(fmaxf(sx, 0.0f), (float)(Ws - 1));
    int y0 = (int)floorf(sy), x0 = (int)floorf(sx);
    float fy = sy - (float)y0, fx = sx - (float)x0;
    int y1 = min(y0 + 1, Hs - 1), x1 = min(x0 + 1, Ws - 1);
    const float* sp = src + ((long long)b * CH + c) * ((long long)Hs * Ws);
    float v00 = sp[(long long)y0 * Ws + x0], v01 = sp[(long long)y0 * Ws + x1];
    float v10 = sp[(long long)y1 * Ws + x0], v11 = sp[(long long)y1 * Ws + x1];
    dst[tid] = (v00 * (1.f - fx) * (1.f - fy) + v01 * fx * (1.f - fy)
              + v10 * (1.f - fx) * fy        + v11 * fx * fy) * scale;
}

// Bilinear resize f32 (single-channel) -> one packed-f16 channel slot.
__global__ void __launch_bounds__(256)
resize_to_pk_kernel(const float* __restrict__ src, _Float16* __restrict__ dstpk,
                    int Bn, int Hs, int Ws, int Hd, int Wd, int Cpad, int cOff)
{
    long long total = (long long)Bn * Hd * Wd;
    long long tid = (long long)blockIdx.x * blockDim.x + threadIdx.x;
    if (tid >= total) return;
    int xd = (int)(tid % Wd);
    int yd = (int)((tid / Wd) % Hd);
    int b  = (int)(tid / ((long long)Wd * Hd));
    float sy = ((float)yd + 0.5f) * ((float)Hs / (float)Hd) - 0.5f;
    float sx = ((float)xd + 0.5f) * ((float)Ws / (float)Wd) - 0.5f;
    sy = fminf(fmaxf(sy, 0.0f), (float)(Hs - 1));
    sx = fminf(fmaxf(sx, 0.0f), (float)(Ws - 1));
    int y0 = (int)floorf(sy), x0 = (int)floorf(sx);
    float fy = sy - (float)y0, fx = sx - (float)x0;
    int y1 = min(y0 + 1, Hs - 1), x1 = min(x0 + 1, Ws - 1);
    const float* sp = src + (long long)b * Hs * Ws;
    float v00 = sp[(long long)y0 * Ws + x0], v01 = sp[(long long)y0 * Ws + x1];
    float v10 = sp[(long long)y1 * Ws + x0], v11 = sp[(long long)y1 * Ws + x1];
    float v = v00 * (1.f - fx) * (1.f - fy) + v01 * fx * (1.f - fy)
            + v10 * (1.f - fx) * fy        + v11 * fx * fy;
    dstpk[tid * Cpad + cOff] = (_Float16)v;
}

// warp (align_corners=True grid_sample, border clamp) -> packed f16 channels
__global__ void __launch_bounds__(256)
warp_to_pk_kernel(const float* __restrict__ feat, const float* __restrict__ flow,
                  _Float16* __restrict__ dstpk, int Bn, int C, int Hf, int Wf,
                  int Cpad, int cOff)
{
    long long HW = (long long)Hf * Wf;
    long long total = (long long)Bn * C * HW;
    long long tid = (long long)blockIdx.x * blockDim.x + threadIdx.x;
    if (tid >= total) return;
    int xp = (int)(tid % Wf);
    int yp = (int)((tid / Wf) % Hf);
    int c  = (int)((tid / HW) % C);
    int b  = (int)(tid / ((long long)C * HW));
    long long p = (long long)yp * Wf + xp;
    float fx = flow[((long long)b * 2 + 0) * HW + p];
    float fy = flow[((long long)b * 2 + 1) * HW + p];
    float sx = fminf(fmaxf((float)xp + fx, 0.0f), (float)(Wf - 1));
    float sy = fminf(fmaxf((float)yp + fy, 0.0f), (float)(Hf - 1));
    int x0 = (int)floorf(sx), y0 = (int)floorf(sy);
    float wx = sx - (float)x0, wy = sy - (float)y0;
    int x1 = min(x0 + 1, Wf - 1), y1 = min(y0 + 1, Hf - 1);
    const float* fp = feat + ((long long)b * C + c) * HW;
    float v00 = fp[(long long)y0 * Wf + x0], v01 = fp[(long long)y0 * Wf + x1];
    float v10 = fp[(long long)y1 * Wf + x0], v11 = fp[(long long)y1 * Wf + x1];
    float v = v00 * (1.f - wx) * (1.f - wy) + v01 * wx * (1.f - wy)
            + v10 * (1.f - wx) * wy        + v11 * wx * wy;
    dstpk[((long long)b * HW + p) * Cpad + cOff + c] = (_Float16)v;
}

// fill packed channels [cStart, Cpad): tmap value at cStart, zeros beyond
__global__ void __launch_bounds__(256)
fill_pk_tail_kernel(_Float16* __restrict__ dstpk, long long Npix,
                    int Cpad, int cStart, float value)
{
    int nfill = Cpad - cStart;
    long long total = Npix * nfill;
    long long tid = (long long)blockIdx.x * blockDim.x + threadIdx.x;
    if (tid >= total) return;
    int ci = (int)(tid % nfill);
    long long n = tid / nfill;
    dstpk[n * Cpad + cStart + ci] = (ci == 0) ? (_Float16)value : (_Float16)0.0f;
}

// source = f1w*(1-T) + f2w*T from fused packed channels [0,C) and [C,2C)
__global__ void __launch_bounds__(256)
source_pk_kernel(const _Float16* __restrict__ fusedpk, _Float16* __restrict__ srcpk,
                 long long Npix, int C, int CpadIn, float t)
{
    long long total = Npix * C;
    long long tid = (long long)blockIdx.x * blockDim.x + threadIdx.x;
    if (tid >= total) return;
    int c = (int)(tid % C);
    long long n = tid / C;
    float a = (float)fusedpk[n * CpadIn + c];
    float d = (float)fusedpk[n * CpadIn + C + c];
    srcpk[n * C + c] = (_Float16)(a * (1.0f - t) + d * t);
}

// GroupNorm on packed f16: one block per (batch, group)
__global__ void __launch_bounds__(256)
groupnorm_pk_kernel(const _Float16* __restrict__ xpk, const float* __restrict__ gamma,
                    const float* __restrict__ beta, _Float16* __restrict__ outpk,
                    int Bn, int C, long long HW, int groups)
{
    __shared__ float rs[256];
    __shared__ float rq[256];
    const int bg = blockIdx.x;
    const int b = bg / groups, g = bg % groups;
    const int cpg = C / groups;
    const long long n = (long long)cpg * HW;
    const _Float16* base = xpk + (long long)b * HW * C + g * cpg;

    float s = 0.0f, q = 0.0f;
    for (long long i = threadIdx.x; i < n; i += blockDim.x) {
        long long pix = i / cpg;
        int c = (int)(i - pix * cpg);
        float v = (float)base[pix * C + c];
        s += v; q += v * v;
    }
    rs[threadIdx.x] = s; rq[threadIdx.x] = q;
    __syncthreads();
    for (int st = 128; st > 0; st >>= 1) {
        if (threadIdx.x < st) {
            rs[threadIdx.x] += rs[threadIdx.x + st];
            rq[threadIdx.x] += rq[threadIdx.x + st];
        }
        __syncthreads();
    }
    const float mean = rs[0] / (float)n;
    const float var  = rq[0] / (float)n - mean * mean;
    const float inv  = rsqrtf(var + 1e-5f);
    _Float16* obase = outpk + (long long)b * HW * C + g * cpg;
    for (long long i = threadIdx.x; i < n; i += blockDim.x) {
        long long pix = i / cpg;
        int c = (int)(i - pix * cpg);
        float v = (float)base[pix * C + c];
        obase[pix * C + c] = (_Float16)((v - mean) * inv * gamma[g * cpg + c] + beta[g * cpg + c]);
    }
}

// softmax over 9 points: att (B, NH*9, HW) f32, in place
__global__ void __launch_bounds__(256)
softmax9_kernel(float* __restrict__ att, int Bn, int NHh, long long HW)
{
    long long total = (long long)Bn * NHh * HW;
    long long tid = (long long)blockIdx.x * blockDim.x + threadIdx.x;
    if (tid >= total) return;
    long long p = tid % HW;
    int nh = (int)((tid / HW) % NHh);
    int b  = (int)(tid / ((long long)NHh * HW));
    float* base = att + (((long long)b * NHh + nh) * 9) * HW + p;
    float m = -3.4e38f;
    float v[9];
#pragma unroll
    for (int i = 0; i < 9; ++i) { v[i] = base[(long long)i * HW]; m = fmaxf(m, v[i]); }
    float s = 0.0f;
#pragma unroll
    for (int i = 0; i < 9; ++i) { v[i] = expf(v[i] - m); s += v[i]; }
    float inv = 1.0f / s;
#pragma unroll
    for (int i = 0; i < 9; ++i) base[(long long)i * HW] = v[i] * inv;
}

// deformable aggregation: vals f32 NCHW planes -> packed f16 output
__global__ void __launch_bounds__(256)
deform_agg_kernel(const float* __restrict__ vals, const float* __restrict__ off,
                  const float* __restrict__ att, _Float16* __restrict__ outpk,
                  int Bn, int C, int Hh, int Ww)
{
    long long HW = (long long)Hh * Ww;
    long long total = (long long)Bn * C * HW;
    long long tid = (long long)blockIdx.x * blockDim.x + threadIdx.x;
    if (tid >= total) return;
    long long p = tid % HW;
    int c = (int)((tid / HW) % C);
    int b = (int)(tid / ((long long)C * HW));
    int xp = (int)(p % Ww), yp = (int)(p / Ww);
    int nh = c >> 5;                        // 32 channels per head
    const float* vp = vals + ((long long)b * C + c) * HW;
    const float xsc = (float)Ww / (float)(Ww - 1);
    const float ysc = (float)Hh / (float)(Hh - 1);
    float accum = 0.0f;
#pragma unroll
    for (int pt = 0; pt < 9; ++pt) {
        int ch2 = (nh * 9 + pt) * 2;
        float ox = off[((long long)b * 72 + ch2 + 0) * HW + p];
        float oy = off[((long long)b * 72 + ch2 + 1) * HW + p];
        float a  = att[((long long)b * 36 + nh * 9 + pt) * HW + p];
        float xs = ((float)xp + ox) * xsc - 0.5f;
        float ys = ((float)yp + oy) * ysc - 0.5f;
        xs = fminf(fmaxf(xs, 0.0f), (float)(Ww - 1));
        ys = fminf(fmaxf(ys, 0.0f), (float)(Hh - 1));
        int x0 = (int)floorf(xs), y0 = (int)floorf(ys);
        float wx = xs - (float)x0, wy = ys - (float)y0;
        int x1 = min(x0 + 1, Ww - 1), y1 = min(y0 + 1, Hh - 1);
        float v00 = vp[(long long)y0 * Ww + x0], v01 = vp[(long long)y0 * Ww + x1];
        float v10 = vp[(long long)y1 * Ww + x0], v11 = vp[(long long)y1 * Ww + x1];
        accum += a * (v00 * (1.f - wx) * (1.f - wy) + v01 * wx * (1.f - wy)
                    + v10 * (1.f - wx) * wy        + v11 * wx * wy);
    }
    outpk[((long long)b * HW + p) * C + c] = (_Float16)accum;
}

// 4x bilinear upsample on packed f16, vectorized 8 channels per thread
__global__ void __launch_bounds__(256)
upsample_pk_kernel(const _Float16* __restrict__ in, _Float16* __restrict__ out,
                   int Bn, int Cpad, int Hs, int Ws, int Hd, int Wd)
{
    const int C8 = Cpad >> 3;
    long long total = (long long)Bn * Hd * Wd * C8;
    long long tid = (long long)blockIdx.x * blockDim.x + threadIdx.x;
    if (tid >= total) return;
    int c8 = (int)(tid % C8);
    long long t = tid / C8;
    int xd = (int)(t % Wd);
    int yd = (int)((t / Wd) % Hd);
    int b  = (int)(t / ((long long)Wd * Hd));
    float sy = ((float)yd + 0.5f) * ((float)Hs / (float)Hd) - 0.5f;
    float sx = ((float)xd + 0.5f) * ((float)Ws / (float)Wd) - 0.5f;
    sy = fminf(fmaxf(sy, 0.0f), (float)(Hs - 1));
    sx = fminf(fmaxf(sx, 0.0f), (float)(Ws - 1));
    int y0 = (int)floorf(sy), x0 = (int)floorf(sx);
    float fy = sy - (float)y0, fx = sx - (float)x0;
    int y1 = min(y0 + 1, Hs - 1), x1 = min(x0 + 1, Ws - 1);
    const long long rowb = (long long)b * Hs * Ws;
    const v8h p00 = *(const v8h*)(in + (rowb + (long long)y0 * Ws + x0) * Cpad + c8 * 8);
    const v8h p01 = *(const v8h*)(in + (rowb + (long long)y0 * Ws + x1) * Cpad + c8 * 8);
    const v8h p10 = *(const v8h*)(in + (rowb + (long long)y1 * Ws + x0) * Cpad + c8 * 8);
    const v8h p11 = *(const v8h*)(in + (rowb + (long long)y1 * Ws + x1) * Cpad + c8 * 8);
    v8h ov;
#pragma unroll
    for (int j = 0; j < 8; ++j) {
        float v = (float)p00[j] * (1.f - fx) * (1.f - fy) + (float)p01[j] * fx * (1.f - fy)
                + (float)p10[j] * (1.f - fx) * fy        + (float)p11[j] * fx * fy;
        ov[j] = (_Float16)v;
    }
    *(v8h*)(out + (((long long)b * Hd + yd) * Wd + xd) * Cpad + c8 * 8) = ov;
}

// ---------------------------------------------------------------------------
// Host orchestration
// ---------------------------------------------------------------------------
static inline unsigned blocks_for(long long n) { return (unsigned)((n + 255) / 256); }

static inline void conv_pk(hipStream_t s, const _Float16* xpk, const _Float16* apk,
                           const float* bias, const _Float16* respk,
                           _Float16* outpk, float* out32,
                           int Bn, int CinPad, int Hh, int Ww,
                           int Cout, int CoutPad, int Ksz, int groups, int cg, int act)
{
    long long Np = (long long)Bn * Hh * Ww;
    dim3 grid((unsigned)((Np + 127) / 128), (unsigned)((Cout + 31) / 32), 1);
    wmma_conv_pk<<<grid, dim3(256), 0, s>>>(xpk, apk, bias, respk, outpk, out32,
                                            Bn, CinPad, Hh, Ww, Cout, CoutPad,
                                            Ksz, groups, cg, act);
}

static inline long long apack_halves(int Cout, int cg, int KK)
{
    int Cpg = (cg + 31) & ~31;
    long long S = (long long)(KK * Cpg) >> 5;
    long long nMt = (Cout + 15) >> 4;
    return S * nMt * 32 * 16;
}

static inline void pack_a(hipStream_t s, const float* w, _Float16* apk,
                          int Cout, int cg, int KK)
{
    int Cpg = (cg + 31) & ~31;
    int S = (KK * Cpg) >> 5;
    int nMt = (Cout + 15) >> 4;
    long long threads = (long long)S * nMt * 32;
    pack_a_kernel<<<blocks_for(threads), 256, 0, s>>>(w, apk, Cout, cg, Cpg, KK, nMt, S);
}

extern "C" void kernel_launch(void* const* d_in, const int* in_sizes, int n_in,
                              void* d_out, int out_size, void* d_ws, size_t ws_size,
                              hipStream_t stream)
{
    (void)in_sizes; (void)n_in; (void)out_size; (void)ws_size;
    const int Bn = 2, C = 128, FH = 96, FW = 160, H = 384, W = 640;
    const int NH = 4;
    const float T = 0.5f;
    const long long HWf = (long long)FH * FW;       // 15360
    const long long HWo = (long long)H * W;         // 245760
    const long long Nf = Bn * HWf;                  // 30720 feat pixels
    const long long No = Bn * HWo;                  // 491520 out pixels
    const int CF = 288;                             // fused channels padded (259->288)

    const float* feat1    = (const float*)d_in[0];
    const float* feat2    = (const float*)d_in[1];
    const float* flow_fwd = (const float*)d_in[2];
    const float* flow_bwd = (const float*)d_in[3];
    const float* edge1    = (const float*)d_in[4];
    const float* edge2    = (const float*)d_in[5];
    const float* fuse_w0  = (const float*)d_in[6];
    const float* fuse_b0  = (const float*)d_in[7];
    const float* fuse_w1  = (const float*)d_in[8];
    const float* fuse_b1  = (const float*)d_in[9];
    const float* gn_g     = (const float*)d_in[10];
    const float* gn_b     = (const float*)d_in[11];
    const float* off_w0   = (const float*)d_in[12];
    const float* off_b0   = (const float*)d_in[13];
    const float* off_w1   = (const float*)d_in[14];
    const float* off_b1   = (const float*)d_in[15];
    const float* aw_w0    = (const float*)d_in[16];
    const float* aw_b0    = (const float*)d_in[17];
    const float* aw_w1    = (const float*)d_in[18];
    const float* aw_b1    = (const float*)d_in[19];
    const float* val_w    = (const float*)d_in[20];
    const float* val_b    = (const float*)d_in[21];
    const float* out_w    = (const float*)d_in[22];
    const float* out_b    = (const float*)d_in[23];
    const float* ffn_w0   = (const float*)d_in[24];
    const float* ffn_b0   = (const float*)d_in[25];
    const float* ffn_w1   = (const float*)d_in[26];
    const float* ffn_b1   = (const float*)d_in[27];
    const float* dec_w0   = (const float*)d_in[28];
    const float* dec_b0   = (const float*)d_in[29];
    const float* dec_w1   = (const float*)d_in[30];
    const float* dec_b1   = (const float*)d_in[31];
    const float* dec_w2   = (const float*)d_in[32];
    const float* dec_b2   = (const float*)d_in[33];
    float* outp = (float*)d_out;

    // ---- workspace: f32 region first, then f16 region ----
    float* f32ws = (float*)d_ws;
    long long fo = 0;
    float* FLOW1 = f32ws + fo; fo += Bn * 2 * HWf;
    float* FLOW2 = f32ws + fo; fo += Bn * 2 * HWf;
    float* OFF   = f32ws + fo; fo += Bn * 72 * HWf;
    float* ATT   = f32ws + fo; fo += Bn * 36 * HWf;
    float* VALS  = f32ws + fo; fo += Bn * C * HWf;
    fo = (fo + 7) & ~7LL;                           // 32B align the f16 region

    _Float16* h16 = (_Float16*)(f32ws + fo);
    long long ho = 0;
    _Float16* FUSEDpk = h16 + ho; ho += Nf * CF;        // warped feats + edges + tmap
    _Float16* T0pk    = h16 + ho; ho += Nf * C;
    _Float16* Xpk     = h16 + ho; ho += Nf * C;
    _Float16* GNpk    = h16 + ho; ho += Nf * C;
    _Float16* TMP1pk  = h16 + ho; ho += Nf * C;         // off_t / attn_t
    _Float16* SRCpk   = h16 + ho; ho += Nf * C;
    _Float16* AGGpk   = h16 + ho; ho += Nf * C;
    _Float16* XATTNpk = h16 + ho; ho += Nf * C;
    _Float16* FFHpk   = h16 + ho; ho += Nf * 2 * C;
    _Float16* X2pk    = h16 + ho; ho += Nf * C;
    _Float16* XUpk    = h16 + ho; ho += No * C;         // 4x upsampled x2
    _Float16* D0pk    = h16 + ho; ho += No * 64;
    _Float16* D1pk    = h16 + ho; ho += No * 32;
    // A-fragment weight packs
    _Float16* A_fuse0 = h16 + ho; ho += apack_halves(C, 259, 9);
    _Float16* A_fuse1 = h16 + ho; ho += apack_halves(C, C, 9);
    _Float16* A_off0  = h16 + ho; ho += apack_halves(C, 32, 9);
    _Float16* A_off1  = h16 + ho; ho += apack_halves(72, C, 1);
    _Float16* A_aw0   = h16 + ho; ho += apack_halves(C, 32, 9);
    _Float16* A_aw1   = h16 + ho; ho += apack_halves(36, C, 1);
    _Float16* A_val   = h16 + ho; ho += apack_halves(C, C, 1);
    _Float16* A_out   = h16 + ho; ho += apack_halves(C, C, 1);
    _Float16* A_ffn0  = h16 + ho; ho += apack_halves(2 * C, C, 1);
    _Float16* A_ffn1  = h16 + ho; ho += apack_halves(C, 2 * C, 1);
    _Float16* A_dec0  = h16 + ho; ho += apack_halves(64, C, 9);
    _Float16* A_dec1  = h16 + ho; ho += apack_halves(32, 64, 9);
    _Float16* A_dec2  = h16 + ho; ho += apack_halves(3, 32, 9);

    // ---- weight pre-packs (deterministic, every call) ----
    pack_a(stream, fuse_w0, A_fuse0, C, 259, 9);
    pack_a(stream, fuse_w1, A_fuse1, C, C, 9);
    pack_a(stream, off_w0,  A_off0,  C, 32, 9);
    pack_a(stream, off_w1,  A_off1,  72, C, 1);
    pack_a(stream, aw_w0,   A_aw0,   C, 32, 9);
    pack_a(stream, aw_w1,   A_aw1,   36, C, 1);
    pack_a(stream, val_w,   A_val,   C, C, 1);
    pack_a(stream, out_w,   A_out,   C, C, 1);
    pack_a(stream, ffn_w0,  A_ffn0,  2 * C, C, 1);
    pack_a(stream, ffn_w1,  A_ffn1,  C, 2 * C, 1);
    pack_a(stream, dec_w0,  A_dec0,  64, C, 9);
    pack_a(stream, dec_w1,  A_dec1,  32, 64, 9);
    pack_a(stream, dec_w2,  A_dec2,  3, 32, 9);

    // ---- flows: resize to feat res, fold flow*T and *0.25 scaling ----
    resize_f32_kernel<<<blocks_for(Bn * 2 * HWf), 256, 0, stream>>>(
        flow_fwd, FLOW1, Bn, 2, H, W, FH, FW, T * ((float)FW / W));
    resize_f32_kernel<<<blocks_for(Bn * 2 * HWf), 256, 0, stream>>>(
        flow_bwd, FLOW2, Bn, 2, H, W, FH, FW, (1.0f - T) * ((float)FW / W));

    // ---- fused packed input: warps, edges, tmap, zero pad ----
    warp_to_pk_kernel<<<blocks_for(Nf * C), 256, 0, stream>>>(
        feat1, FLOW1, FUSEDpk, Bn, C, FH, FW, CF, 0);
    warp_to_pk_kernel<<<blocks_for(Nf * C), 256, 0, stream>>>(
        feat2, FLOW2, FUSEDpk, Bn, C, FH, FW, CF, 128);
    resize_to_pk_kernel<<<blocks_for(Nf), 256, 0, stream>>>(
        edge1, FUSEDpk, Bn, H, W, FH, FW, CF, 256);
    resize_to_pk_kernel<<<blocks_for(Nf), 256, 0, stream>>>(
        edge2, FUSEDpk, Bn, H, W, FH, FW, CF, 257);
    fill_pk_tail_kernel<<<blocks_for(Nf * (CF - 258)), 256, 0, stream>>>(
        FUSEDpk, Nf, CF, 258, T);

    // ---- source = f1w*(1-T) + f2w*T ----
    source_pk_kernel<<<blocks_for(Nf * C), 256, 0, stream>>>(
        FUSEDpk, SRCpk, Nf, C, CF, T);

    // ---- fuse convs ----
    conv_pk(stream, FUSEDpk, A_fuse0, fuse_b0, nullptr, T0pk, nullptr,
            Bn, CF, FH, FW, C, C, 3, 1, 259, ACT_LRELU);
    conv_pk(stream, T0pk, A_fuse1, fuse_b1, nullptr, Xpk, nullptr,
            Bn, C, FH, FW, C, C, 3, 1, C, ACT_LRELU);

    // ---- group norm ----
    groupnorm_pk_kernel<<<Bn * 8, 256, 0, stream>>>(Xpk, gn_g, gn_b, GNpk, Bn, C, HWf, 8);

    // ---- deformable attention ----
    conv_pk(stream, GNpk, A_off0, off_b0, nullptr, TMP1pk, nullptr,
            Bn, C, FH, FW, C, C, 3, NH, 32, ACT_LRELU);
    conv_pk(stream, TMP1pk, A_off1, off_b1, nullptr, nullptr, OFF,
            Bn, C, FH, FW, 72, 0, 1, 1, C, ACT_NONE);
    conv_pk(stream, GNpk, A_aw0, aw_b0, nullptr, TMP1pk, nullptr,
            Bn, C, FH, FW, C, C, 3, NH, 32, ACT_LRELU);
    conv_pk(stream, TMP1pk, A_aw1, aw_b1, nullptr, nullptr, ATT,
            Bn, C, FH, FW, 36, 0, 1, 1, C, ACT_NONE);
    softmax9_kernel<<<blocks_for(Bn * NH * HWf), 256, 0, stream>>>(ATT, Bn, NH, HWf);
    conv_pk(stream, SRCpk, A_val, val_b, nullptr, nullptr, VALS,
            Bn, C, FH, FW, C, C, 1, 1, C, ACT_NONE);
    deform_agg_kernel<<<blocks_for(Nf * C), 256, 0, stream>>>(
        VALS, OFF, ATT, AGGpk, Bn, C, FH, FW);
    conv_pk(stream, AGGpk, A_out, out_b, Xpk, XATTNpk, nullptr,   // +x residual
            Bn, C, FH, FW, C, C, 1, 1, C, ACT_NONE);

    // ---- FFN ----
    conv_pk(stream, XATTNpk, A_ffn0, ffn_b0, nullptr, FFHpk, nullptr,
            Bn, C, FH, FW, 2 * C, 2 * C, 1, 1, C, ACT_GELU);
    conv_pk(stream, FFHpk, A_ffn1, ffn_b1, XATTNpk, X2pk, nullptr, // +x_attn residual
            Bn, 2 * C, FH, FW, C, C, 1, 1, 2 * C, ACT_NONE);

    // ---- decoder: materialize 4x upsample in packed f16, then 3 convs ----
    upsample_pk_kernel<<<blocks_for(No * (C / 8)), 256, 0, stream>>>(
        X2pk, XUpk, Bn, C, FH, FW, H, W);
    conv_pk(stream, XUpk, A_dec0, dec_b0, nullptr, D0pk, nullptr,
            Bn, C, H, W, 64, 64, 3, 1, C, ACT_LRELU);
    conv_pk(stream, D0pk, A_dec1, dec_b1, nullptr, D1pk, nullptr,
            Bn, 64, H, W, 32, 32, 3, 1, 64, ACT_LRELU);
    conv_pk(stream, D1pk, A_dec2, dec_b2, nullptr, nullptr, outp,
            Bn, 32, H, W, 3, 0, 3, 1, 32, ACT_SIGMOID);
}
